// LaStTemporalPooling_51299089383910
// MI455X (gfx1250) — matrix operand order repaired and checked by reading, not compiled, not run on hardware
//
#include <hip/hip_runtime.h>

typedef __attribute__((ext_vector_type(16))) _Float16 v16h;
typedef __attribute__((ext_vector_type(8)))  _Float16 v8h;
typedef __attribute__((ext_vector_type(8)))  float    v8f;

#define N_   64
#define C_   256
#define S_   30
#define H_   16
#define W_   11
#define TOPK 10
#define EPS_ 1e-6f

#define XSP  31    // Xs row stride (floats)  : 31 coprime 64 -> conflict-free
#define XHS  264   // Xh row stride (halves)  : 132 dwords, 132%64=4 -> lanes spread banks
#define SCP  33    // Sc row stride (floats)  : 33 coprime 64 -> conflict-free top-k scans

// Manual LDS placement: Xh first (WMMA B-frags need small absolute addresses so
// ds_load_b128 can use base-reg + 16-bit immediate offsets -> no address VALU
// between WMMAs -> no WMMA->VALU hazard NOPs). Sc second (<64KB). Xs last
// (accessed via per-thread base pointer, absolute address irrelevant).
#define XH_OFF 0
#define XH_BYTES (32 * XHS * 2)            // 16896
#define SC_OFF XH_BYTES                    // 16896
#define SC_BYTES (C_ * SCP * 4)            // 33792
#define XS_OFF (SC_OFF + SC_BYTES)         // 50688
#define XS_BYTES (C_ * XSP * 4)            // 31744
#define SMEM_BYTES (XS_OFF + XS_BYTES)     // 82432

// ---------------------------------------------------------------------------
// Kernel 0: build the circulant low-pass matrix W[m][k] = g[(m-k) & 255],
// pre-swizzled into the V_WMMA_F32_16X16X32_F16 A-fragment layout (f16).
// Per ISA 7.12.2 (16-bit A 16x32): lane holds row M = lane&15;
// half i holds K = kt*32 + 16*(i>>3) + 8*(lane>>4) + (i&7).
// g[d] = (1/256)*(mask0 + 2*sum_{j=1..127} mask_j cos(2pi j d/256)
//                 + mask128 * (-1)^d), mask_j = exp(-0.5*((j/128)/0.25)^2)
// ---------------------------------------------------------------------------
__global__ void build_wmat(_Float16* __restrict__ aswz) {
  __shared__ float gk[256];
  const int tid = threadIdx.x;
  {
    const int d = tid;
    float acc = 0.0f;
    const float inv_sig = 4.0f;  // 1/0.25
    for (int j = 0; j <= 128; ++j) {
      float fa = (float)j * (1.0f / 128.0f);
      float m  = __expf(-0.5f * fa * fa * inv_sig * inv_sig);
      int   t  = (j * d) & 255;  // exact period-256 argument reduction
      float cw = __cosf(6.283185307179586f * (float)t * (1.0f / 256.0f));
      float wgt = (j == 0 || j == 128) ? 1.0f : 2.0f;
      acc += wgt * m * cw;
    }
    gk[d] = acc * (1.0f / 256.0f);
  }
  __syncthreads();
  // 65536 f16 entries total; each of 64 blocks writes 1024.
  for (int i = 0; i < 4; ++i) {
    int e    = blockIdx.x * 1024 + i * 256 + tid;
    int elem = e & 15;
    int lane = (e >> 4) & 31;
    int kt   = (e >> 9) & 7;
    int mt   = e >> 12;
    int m = mt * 16 + (lane & 15);
    int k = kt * 32 + ((elem >> 3) << 4) + ((lane >> 4) << 3) + (elem & 7);
    aswz[e] = (_Float16)gk[(m - k) & 255];
  }
}

// ---------------------------------------------------------------------------
// Kernel 1: fused  low-pass GEMM -> stability score -> top-10(s) mean pool
//           + max pool -> sigmoid fusion.  One 8-wave block per (n,h,w).
// ---------------------------------------------------------------------------
__global__ void __launch_bounds__(256)
fused_pool(const float* __restrict__ seqs,
           const float* __restrict__ fusion_logit,
           const _Float16* __restrict__ aswz,
           float* __restrict__ out) {
  __shared__ __align__(32) unsigned char smem[SMEM_BYTES];
  _Float16* Xh = (_Float16*)(smem + XH_OFF);  // f16 tokens [s][c], s padded->32 w/ 0
  float*    Sc = (float*)(smem + SC_OFF);     // scores [c][s]
  float*    Xs = (float*)(smem + XS_OFF);     // f32 tokens [c][s]

  const int b = blockIdx.x;
  const int w = b % W_;
  const int h = (b / W_) % H_;
  const int n = b / (W_ * H_);
  const int tid  = threadIdx.x;
  const int lane = tid & 31;
  const int wave = tid >> 5;

  // ---- stage the 256(c) x 30(s) slab for this (n,h,w) ----
  const float* basep = seqs + (size_t)n * (C_ * S_ * H_ * W_) + h * W_ + w;
  {
    const int c = tid;
    for (int s = 0; s < S_; ++s) {
      float v = basep[(size_t)(c * S_ + s) * (H_ * W_)];
      Xs[c * XSP + s] = v;
      Xh[s * XHS + c] = (_Float16)v;
    }
    Xh[30 * XHS + c] = (_Float16)0.0f;  // zero-pad s=30,31 so WMMA needs no mask
    Xh[31 * XHS + c] = (_Float16)0.0f;
  }
  __syncthreads();

  // ---- GEMM: L[c][s] = sum_k W[c][k] * X[k][s] via v_wmma_f32_16x16x32_f16 ----
  // 8 waves, each owns 2 M-tiles; loop 2 N-tiles x 8 K-tiles, B shared per K.
  const int mt0 = wave * 2;
  for (int nt = 0; nt < 2; ++nt) {
    v8f acc0 = {}; v8f acc1 = {};
    const int s = nt * 16 + (lane & 15);                   // B-frag column N
    const _Float16* bptr = &Xh[s * XHS + ((lane >> 4) << 3)];
    #pragma unroll
    for (int kt = 0; kt < 8; ++kt) {
      // K (c) = 8h+0..7 and 16+8h+0..7 as two 16B-aligned halves -> ds_load_b128 pair,
      // base register + immediate offsets (kt*64 bytes)
      v8h blo = *(const v8h*)(bptr + kt * 32);
      v8h bhi = *(const v8h*)(bptr + kt * 32 + 16);
      v16h bfrag = __builtin_shufflevector(blo, bhi,
          0, 1, 2, 3, 4, 5, 6, 7, 8, 9, 10, 11, 12, 13, 14, 15);
      v16h a0 = *(const v16h*)(aswz + ((((mt0 + 0) * 8 + kt) * 32 + lane) << 4));
      acc0 = __builtin_amdgcn_wmma_f32_16x16x32_f16(
          false, a0, false, bfrag, (short)0, acc0, false, false);
      v16h a1 = *(const v16h*)(aswz + ((((mt0 + 1) * 8 + kt) * 32 + lane) << 4));
      acc1 = __builtin_amdgcn_wmma_f32_16x16x32_f16(
          false, a1, false, bfrag, (short)0, acc1, false, false);
    }
    // D layout: VGPR r -> M = r + 8*(lane>>4), N = lane&15
    if (s < S_) {
      const int cg = (lane >> 4) << 3;
      const float* x0p = &Xs[((mt0 + 0) * 16 + cg) * XSP + s];
      const float* x1p = &Xs[((mt0 + 1) * 16 + cg) * XSP + s];
      float x0[8], x1[8];
      #pragma unroll
      for (int r = 0; r < 8; ++r) {      // batched gather: one clause, one wait
        x0[r] = x0p[r * XSP];
        x1[r] = x1p[r * XSP];
      }
      float* s0p = &Sc[((mt0 + 0) * 16 + cg) * SCP + s];
      float* s1p = &Sc[((mt0 + 1) * 16 + cg) * SCP + s];
      #pragma unroll
      for (int r = 0; r < 8; ++r) {
        float L0 = acc0[r];
        s0p[r * SCP] = fabsf(L0) / (fabsf(L0 - x0[r]) + EPS_);
        float L1 = acc1[r];
        s1p[r * SCP] = fabsf(L1) / (fabsf(L1 - x1[r]) + EPS_);
      }
    }
  }
  __syncthreads();

  // ---- per-channel: top-10 over s (first-index tie-break), mean, max, fuse ----
  {
    const int c = tid;
    const float* sc = &Sc[c * SCP];
    const float* xr = &Xs[c * XSP];

    // Pull the whole row into registers once (clause-able ds_load_b32 batch);
    // selection then runs entirely in VALU with no LDS round-trips.
    float scv[S_], xv[S_];
    #pragma unroll
    for (int s = 0; s < S_; ++s) { scv[s] = sc[s]; xv[s] = xr[s]; }

    float mx = xv[0];
    #pragma unroll
    for (int s = 1; s < S_; ++s) mx = fmaxf(mx, xv[s]);

    unsigned taken = 0u;
    float sum = 0.0f;
    for (int k = 0; k < TOPK; ++k) {
      float best = -__builtin_inff();
      float bx   = 0.0f;
      int   bi   = 0;
      #pragma unroll
      for (int s = 0; s < S_; ++s) {
        bool take = (((taken >> s) & 1u) == 0u) && (scv[s] > best);
        best = take ? scv[s] : best;
        bx   = take ? xv[s]  : bx;
        bi   = take ? s      : bi;
      }
      taken |= (1u << bi);
      sum += bx;
    }

    float fl = fusion_logit[0];
    float alpha = 1.0f / (1.0f + __expf(-fl));
    float res = alpha * (sum * (1.0f / (float)TOPK)) + (1.0f - alpha) * mx;
    out[(((size_t)n * C_ + c) * H_ + h) * W_ + w] = res;
  }
}

// ---------------------------------------------------------------------------
extern "C" void kernel_launch(void* const* d_in, const int* in_sizes, int n_in,
                              void* d_out, int out_size, void* d_ws, size_t ws_size,
                              hipStream_t stream) {
  const float* seqs = (const float*)d_in[0];
  const float* fl   = (const float*)d_in[1];
  _Float16* aswz    = (_Float16*)d_ws;          // 128 KB swizzled W matrix
  float* outp       = (float*)d_out;

  build_wmat<<<64, 256, 0, stream>>>(aswz);
  fused_pool<<<N_ * H_ * W_, 256, 0, stream>>>(seqs, fl, aswz, outp);
}
